// KANAdaptiveFusion_7361573945854
// MI455X (gfx1250) — compile-verified, matrix operand order checked by reading, not run on hardware
//
#include <hip/hip_runtime.h>
#include <hip/hip_bf16.h>

// ---------------------------------------------------------------------------
// KAN (B-spline) linear + LayerNorm, fused as one bf16 WMMA GEMM on gfx1250.
//   A[b, i*9+j] = j==0 ? silu(x[b,i]) : Bspline_j-1(x[b,i])      (16384 x 9216)
//   Wc[o, i*9+j] = j==0 ? base_w[o,i] : spline_w[o,i,j-1]*scaler (  512 x 9216)
//   out = LayerNorm(A @ Wc^T)
// ---------------------------------------------------------------------------

typedef __attribute__((ext_vector_type(16))) __bf16 v16bf;
typedef __attribute__((ext_vector_type(8)))  __bf16 v8bf;
typedef __attribute__((ext_vector_type(8)))  float  v8f;

#define B_ROWS   16384
#define IN_F     1024
#define OUT_F    512
#define NCH      9            // silu + 8 spline bases per input feature
#define KDIM     (IN_F * NCH) // 9216
#define BM       32           // rows per block
#define IC       32           // input features per chunk
#define KT       (IC * NCH)   // 288 K per chunk (9 WMMA steps)
#define KPAD     296          // padded LDS row stride (conflict-free b128 reads)

__device__ __forceinline__ unsigned short f2bf(float f) {
  unsigned int u = __float_as_uint(f);
  unsigned int r = (u + 0x7FFFu + ((u >> 16) & 1u)) >> 16;
  return (unsigned short)r;
}

// silu + cubic uniform B-spline bases on knots t_n = -1 + 0.4*(n-3), n=0..11
__device__ __forceinline__ void kan_feats(float x, float f[NCH]) {
  f[0] = x * __frcp_rn(1.0f + __expf(-x));   // silu
  const float h = 0.4f;
  float b[11];
#pragma unroll
  for (int n = 0; n < 11; ++n) {
    float tn  = -1.0f + h * (float)(n - 3);
    float tn1 = tn + h;
    b[n] = (x >= tn && x < tn1) ? 1.0f : 0.0f;
  }
#pragma unroll
  for (int k = 1; k <= 3; ++k) {
    float inv = 1.0f / (h * (float)k);
#pragma unroll
    for (int n = 0; n < 11 - 3; ++n) {     // only first 11-k needed; 8 covers k<=3
      if (n < 11 - k) {
        float tn    = -1.0f + h * (float)(n - 3);
        float left  = (x - tn) * inv;
        float right = ((tn + h * (float)(k + 1)) - x) * inv;
        b[n] = left * b[n] + right * b[n + 1];
      }
    }
    // tail entries for k==1,2 (n in [8, 11-k))
#pragma unroll
    for (int n = 8; n < 10; ++n) {
      if (n < 11 - k) {
        float tn    = -1.0f + h * (float)(n - 3);
        float left  = (x - tn) * inv;
        float right = ((tn + h * (float)(k + 1)) - x) * inv;
        b[n] = left * b[n] + right * b[n + 1];
      }
    }
  }
#pragma unroll
  for (int j = 0; j < 8; ++j) f[1 + j] = b[j];
}

// --------------------------- kernel 1: weight prep -------------------------
__global__ __launch_bounds__(256) void prep_weights(
    const float* __restrict__ bw, const float* __restrict__ sw,
    const float* __restrict__ sc, unsigned short* __restrict__ wbf) {
  int idx = blockIdx.x * 256 + threadIdx.x;      // over OUT_F*IN_F
  if (idx >= OUT_F * IN_F) return;
  int o = idx >> 10, i = idx & (IN_F - 1);
  float s = sc[idx];
  unsigned short* dst = wbf + o * KDIM + i * NCH;
  dst[0] = f2bf(bw[idx]);
  const float* sp = sw + idx * 8;
#pragma unroll
  for (int j = 0; j < 8; ++j) dst[1 + j] = f2bf(sp[j] * s);
}

// ------------------- kernel 2: fused A-gen + WMMA GEMM + LN ----------------
__global__ __launch_bounds__(256) void kan_fused(
    const float* __restrict__ rgb, const float* __restrict__ tac,
    const unsigned short* __restrict__ wbf,
    const float* __restrict__ gamma, const float* __restrict__ beta,
    float* __restrict__ out) {
  __shared__ alignas(16) unsigned short As[BM * KPAD];
  __shared__ float sSum[BM], sSumSq[BM];

  const int tid   = threadIdx.x;
  const int lane  = tid & 31;
  const int wave  = tid >> 5;          // 0..7 -> N slice of 64
  const int laneN = lane & 15;
  const int hi    = lane >> 4;         // 0: K 0-7/16-23, 1: K 8-15/24-31
  const int koff  = hi * 8;
  const int rowBase = blockIdx.x * BM;
  const int nWave   = wave * 64;

  if (tid < BM) { sSum[tid] = 0.0f; sSumSq[tid] = 0.0f; }

  v8f acc[2][4];
#pragma unroll
  for (int mt = 0; mt < 2; ++mt)
#pragma unroll
    for (int nt = 0; nt < 4; ++nt)
#pragma unroll
      for (int r = 0; r < 8; ++r) acc[mt][nt][r] = 0.0f;

  for (int chunk = 0; chunk < IN_F / IC; ++chunk) {
    const int i0 = chunk * IC;
    __syncthreads();  // previous chunk's frag reads complete before overwrite
    // ---- generate 32x288 bf16 A tile: 1024 (row, feat) pairs, 4/thread ----
#pragma unroll
    for (int q = 0; q < 4; ++q) {
      int p  = tid + q * 256;
      int il = p & (IC - 1);
      int r  = p >> 5;
      int i  = i0 + il;
      int b  = rowBase + r;
      float x = (i < 512) ? rgb[b * 512 + i] : tac[b * 512 + (i - 512)];
      float f[NCH];
      kan_feats(x, f);
      unsigned short* dst = As + r * KPAD + il * NCH;
#pragma unroll
      for (int j = 0; j < NCH; ++j) dst[j] = f2bf(f[j]);
    }
    __syncthreads();

    const int kg0 = i0 * NCH;
    // ---- 9 WMMA K-steps over this chunk ----
#pragma unroll
    for (int ks = 0; ks < KT; ks += 32) {
      v16bf afrag[2];
#pragma unroll
      for (int mt = 0; mt < 2; ++mt) {
        const unsigned short* ap = As + (mt * 16 + laneN) * KPAD + ks + koff;
        v8bf alo = *(const v8bf*)(const void*)(ap);
        v8bf ahi = *(const v8bf*)(const void*)(ap + 16);
        afrag[mt] = __builtin_shufflevector(alo, ahi, 0, 1, 2, 3, 4, 5, 6, 7,
                                            8, 9, 10, 11, 12, 13, 14, 15);
      }
#pragma unroll
      for (int nt = 0; nt < 4; ++nt) {
        const unsigned short* bp =
            wbf + (nWave + nt * 16 + laneN) * KDIM + kg0 + ks + koff;
        v8bf blo = *(const v8bf*)(const void*)(bp);
        v8bf bhi = *(const v8bf*)(const void*)(bp + 16);
        v16bf bfrag = __builtin_shufflevector(blo, bhi, 0, 1, 2, 3, 4, 5, 6, 7,
                                              8, 9, 10, 11, 12, 13, 14, 15);
#pragma unroll
        for (int mt = 0; mt < 2; ++mt) {
          acc[mt][nt] = __builtin_amdgcn_wmma_f32_16x16x32_bf16(
              false, afrag[mt], false, bfrag, (short)0, acc[mt][nt], false,
              false);
        }
      }
    }
  }

  // ---------------- fused LayerNorm over N=512 (whole row in block) --------
  __syncthreads();  // sSum/sSumSq zeros visible
#pragma unroll
  for (int mt = 0; mt < 2; ++mt) {
    float s[8], sq[8];
#pragma unroll
    for (int r = 0; r < 8; ++r) { s[r] = 0.0f; sq[r] = 0.0f; }
#pragma unroll
    for (int nt = 0; nt < 4; ++nt)
#pragma unroll
      for (int r = 0; r < 8; ++r) {
        float v = acc[mt][nt][r];
        s[r] += v;
        sq[r] += v * v;
      }
    // butterfly over the 16-lane column group (stays within each wave half)
#pragma unroll
    for (int m = 1; m < 16; m <<= 1) {
#pragma unroll
      for (int r = 0; r < 8; ++r) {
        s[r]  += __shfl_xor(s[r], m, 32);
        sq[r] += __shfl_xor(sq[r], m, 32);
      }
    }
    if (laneN == 0) {
#pragma unroll
      for (int r = 0; r < 8; ++r) {
        int row = mt * 16 + hi * 8 + r;
        atomicAdd(&sSum[row], s[r]);      // ds_add_f32
        atomicAdd(&sSumSq[row], sq[r]);
      }
    }
  }
  __syncthreads();

#pragma unroll
  for (int nt = 0; nt < 4; ++nt) {
    int col = nWave + nt * 16 + laneN;
    float g  = gamma[col];
    float be = beta[col];
#pragma unroll
    for (int mt = 0; mt < 2; ++mt) {
#pragma unroll
      for (int r = 0; r < 8; ++r) {
        int row   = mt * 16 + hi * 8 + r;
        float mu  = sSum[row] * (1.0f / (float)OUT_F);
        float var = sSumSq[row] * (1.0f / (float)OUT_F) - mu * mu;
        float v   = (acc[mt][nt][r] - mu) * rsqrtf(var + 1e-5f);
        out[(rowBase + row) * OUT_F + col] = v * g + be;
      }
    }
  }
}

// ---------------------------------------------------------------------------
extern "C" void kernel_launch(void* const* d_in, const int* in_sizes, int n_in,
                              void* d_out, int out_size, void* d_ws,
                              size_t ws_size, hipStream_t stream) {
  const float* rgb   = (const float*)d_in[0];  // (16384, 512)
  const float* tac   = (const float*)d_in[1];  // (16384, 512)
  const float* bw    = (const float*)d_in[2];  // (512, 1024)
  const float* sw    = (const float*)d_in[3];  // (512, 1024, 8)
  const float* sc    = (const float*)d_in[4];  // (512, 1024)
  const float* gamma = (const float*)d_in[5];  // (512,)
  const float* beta  = (const float*)d_in[6];  // (512,)
  float* outp = (float*)d_out;                 // (16384, 512) fp32

  unsigned short* wbf = (unsigned short*)d_ws;  // 512*9216*2 = 9.44 MB bf16

  prep_weights<<<(OUT_F * IN_F + 255) / 256, 256, 0, stream>>>(bw, sw, sc, wbf);
  kan_fused<<<B_ROWS / BM, 256, 0, stream>>>(rgb, tac, wbf, gamma, beta, outp);
}
